// SpikeNet_55456617726550
// MI455X (gfx1250) — matrix operand order repaired and checked by reading, not compiled, not run on hardware
//
#include <hip/hip_runtime.h>
#include <stddef.h>

// ---------------------------------------------------------------------------
// SNN (RLeaky x3) over T=256 steps on gfx1250 using v_wmma_f32_16x16x32_bf16.
// Per step: 3 fused GEMM+neuron kernels; K-dim concatenates feedforward and
// recurrent inputs so cur + spk@Vw^T is a single WMMA GEMM.
// Weights packed once into WMMA B-fragment order; live in L2 (192MB) forever.
// Inner loop is software-pipelined (ping-pong register buffers, unroll 2)
// with WGP-scope global_prefetch of the weight stream. 64-thread blocks
// (2 waves) spread the fixed 512-wave GEMM across all WGPs. Output records
// use non-temporal stores to keep the L2 weight set unpolluted.
// ---------------------------------------------------------------------------

typedef __bf16 bf16_t;
typedef __attribute__((ext_vector_type(16))) __bf16 v16bf;
typedef __attribute__((ext_vector_type(8)))  __bf16 v8bf;
typedef __attribute__((ext_vector_type(8)))  float  v8f;

#define T_STEPS 256
#define BSZ     256
#define NIN     512
#define NHID    2048
#define NOUT    128

// ---------------------------------------------------------------------------
// Pack [W | V] (both row-major [N][K]) transposed into WMMA B-fragment order.
// Fragment layout for V_WMMA_F32_16X16X32_BF16 B operand (32x16, KxN):
//   lane l: column n = l%16 ; element e (0..15): k_in_tile = (l>=16)*8 + (e&7) + (e>=8)*16
// Storage: dst[((kt*NT + nt)*32 + lane)*16 + e]
// ---------------------------------------------------------------------------
__global__ void pack_w(const float* __restrict__ srcA, int K1,
                       const float* __restrict__ srcB, int K2,
                       int N, bf16_t* __restrict__ dst, int total) {
  int d = blockIdx.x * blockDim.x + threadIdx.x;
  if (d >= total) return;
  int e    = d & 15;
  int lane = (d >> 4) & 31;
  int tile = d >> 9;
  int NT   = N >> 4;
  int kt   = tile / NT;
  int nt   = tile - kt * NT;
  int k    = (kt << 5) + ((lane >> 4) << 3) + (e & 7) + ((e >> 3) << 4);
  int n    = (nt << 4) + (lane & 15);
  float v = (k < K1) ? srcA[(size_t)n * K1 + k]
                     : srcB[(size_t)n * K2 + (k - K1)];
  dst[d] = (bf16_t)v;
}

__global__ void combine_bias(const float* __restrict__ b,
                             const float* __restrict__ vb,
                             float* __restrict__ dst, int n) {
  int i = blockIdx.x * blockDim.x + threadIdx.x;
  if (i < n) dst[i] = b[i] + vb[i];
}

__global__ void zero_u32(unsigned int* __restrict__ p, int n) {
  int i = blockIdx.x * blockDim.x + threadIdx.x;
  if (i < n) p[i] = 0u;
}

// ---------------------------------------------------------------------------
// Fragment load / WMMA helpers
// ---------------------------------------------------------------------------
__device__ __forceinline__ void prefetch_wgp(const void* p) {
  // scope omitted => SCOPE 0 (WGP): pull into all cache levels on miss.
  asm volatile("global_prefetch_b8 %0, off" ::"v"(p));
}

__device__ __forceinline__ void load_b4(const bf16_t* __restrict__ q,
                                        v16bf b[4]) {
#pragma unroll
  for (int j = 0; j < 4; ++j) {
    const bf16_t* p = q + (size_t)j * 512;
    *(v8bf*)&b[j]       = *(const v8bf*)p;
    *((v8bf*)&b[j] + 1) = *(const v8bf*)(p + 8);
  }
}

__device__ __forceinline__ void wmma4(const v16bf& a, const v16bf b[4],
                                      v8f acc[4]) {
#pragma unroll
  for (int j = 0; j < 4; ++j)
    acc[j] = __builtin_amdgcn_wmma_f32_16x16x32_bf16(
        false, a, false, b[j], (short)0, acc[j], false, false);
}

// One K-segment of the GEMM, software-pipelined (requires even ktiles >= 2).
// wp0 points at this segment's first k-tile for this wave's first n-tile+lane.
template <bool F32A>
__device__ __forceinline__ void gemm_seg(const void* __restrict__ abase,
                                         int Kseg, int ktiles,
                                         const bf16_t* __restrict__ wp0,
                                         size_t wpStride, int row, int hsel,
                                         v8f acc[4]) {
  auto loadA = [&](int kt, v16bf& a) {
    if (F32A) {
      const float* p =
          (const float*)abase + (size_t)row * Kseg + (size_t)kt * 32 + hsel * 8;
#pragma unroll
      for (int i = 0; i < 8; ++i) a[i] = (bf16_t)p[i];
#pragma unroll
      for (int i = 0; i < 8; ++i) a[8 + i] = (bf16_t)p[16 + i];
    } else {
      const bf16_t* p = (const bf16_t*)abase + (size_t)row * Kseg +
                        (size_t)kt * 32 + hsel * 8;
      *(v8bf*)&a       = *(const v8bf*)p;
      *((v8bf*)&a + 1) = *(const v8bf*)(p + 16);
    }
  };

  v16bf a0, a1;
  v16bf b0[4], b1[4];
  loadA(0, a0);
  load_b4(wp0, b0);
  int kt = 0;
  for (; kt + 2 < ktiles; kt += 2) {
    // loads for kt+1 issue while WMMAs on kt execute
    loadA(kt + 1, a1);
    load_b4(wp0 + (size_t)(kt + 1) * wpStride, b1);
    // WGP-scope prefetch of the k+5 B tiles (all 4 n-tiles of this wave)
#pragma unroll
    for (int j = 0; j < 4; ++j)
      prefetch_wgp(wp0 + (size_t)(kt + 5) * wpStride + (size_t)j * 512);
    wmma4(a0, b0, acc);
    // loads for kt+2 issue while WMMAs on kt+1 execute
    loadA(kt + 2, a0);
    load_b4(wp0 + (size_t)(kt + 2) * wpStride, b0);
    wmma4(a1, b1, acc);
  }
  // tail: kt == ktiles-2
  loadA(kt + 1, a1);
  load_b4(wp0 + (size_t)(kt + 1) * wpStride, b1);
  wmma4(a0, b0, acc);
  wmma4(a1, b1, acc);
}

// ---------------------------------------------------------------------------
// Fused GEMM + RLeaky neuron update.
//   state[b,n] = clip(beta[n],0,1)*m[b,n] + sum_k A[b,k]*Wcat[n,k] + bias[n]
//   MODE 0 (layer1): A = [x_t (f32) | s1_prev], reset-subtract
//   MODE 1 (layer2): A = [s1_new | s2_prev], reset-subtract, also emit m2 bf16
//   MODE 2 (out)   : A = [m2 (bf16) | so_prev], no subtract, emit spk/mem recs
// Block: 64 thr (2 waves), computes 16(M) x 128(N); wave w covers 64 cols
// (4 WMMA tiles, A fragment reused across them). Fine-grained grid spreads
// the fixed wave count across all WGPs.
// ---------------------------------------------------------------------------
template <int MODE>
__global__ __launch_bounds__(64)
void rleaky_gemm(const void* __restrict__ A1p, int K1,
                 const bf16_t* __restrict__ A2, int K2,
                 const bf16_t* __restrict__ Wp,
                 const float* __restrict__ bias,
                 const float* __restrict__ beta,
                 const float* __restrict__ thr,
                 float* __restrict__ mstate,
                 bf16_t* __restrict__ sbf_out,
                 bf16_t* __restrict__ mbf_out,
                 float* __restrict__ spk_rec,
                 float* __restrict__ mem_rec,
                 int N) {
  const int lane  = threadIdx.x & 31;
  const int nHalf = threadIdx.x >> 5;  // wave id: which 64-col half
  const int nBase = blockIdx.y * 128 + nHalf * 64;
  const int row   = blockIdx.x * 16 + (lane & 15);
  const int hsel  = lane >> 4;  // selects K-halves in A fragment
  const int NT16  = N >> 4;

  v8f acc[4] = {};

  const size_t wpStride = (size_t)NT16 * 512;  // halfs per k-tile row of tiles
  const bf16_t* wpCol = Wp + (size_t)(nBase >> 4) * 512 + (size_t)lane * 16;

  // segment 1: feedforward input (f32 for layer1, bf16 otherwise)
  gemm_seg<MODE == 0>(A1p, K1, K1 >> 5, wpCol, wpStride, row, hsel, acc);
  // segment 2: recurrent input (always bf16)
  gemm_seg<false>(A2, K2, K2 >> 5, wpCol + (size_t)(K1 >> 5) * wpStride,
                  wpStride, row, hsel, acc);

  // ---- Fused RLeaky epilogue (C layout: elem i -> row 8*(lane/16)+i, col lane%16)
  const float thrv = thr[0];
#pragma unroll
  for (int j = 0; j < 4; ++j) {
    const int n = nBase + j * 16 + (lane & 15);
    const float betac = fminf(fmaxf(beta[n], 0.0f), 1.0f);
    const float bi = bias[n];
#pragma unroll
    for (int i = 0; i < 8; ++i) {
      const int brow = blockIdx.x * 16 + hsel * 8 + i;
      const size_t idx = (size_t)brow * N + n;
      const float mprev = mstate[idx];
      const float state = betac * mprev + acc[j][i] + bi;
      float mnew;
      if (MODE == 2) {
        mnew = state;                                   // no reset-subtract
      } else {
        const float reset = (mprev - thrv > 0.0f) ? thrv : 0.0f;
        mnew = state - reset;                           // subtract (detached)
      }
      const float snew = (mnew - thrv > 0.0f) ? 1.0f : 0.0f;
      mstate[idx]  = mnew;
      sbf_out[idx] = (bf16_t)snew;
      if (MODE == 1) mbf_out[idx] = (bf16_t)mnew;       // feeds output GEMM
      if (MODE == 2) {
        // write-once record streams: non-temporal, keep weights in L2
        __builtin_nontemporal_store(snew, &spk_rec[idx]);
        __builtin_nontemporal_store(mnew, &mem_rec[idx]);
      }
    }
  }
}

// ---------------------------------------------------------------------------
extern "C" void kernel_launch(void* const* d_in, const int* in_sizes, int n_in,
                              void* d_out, int out_size, void* d_ws,
                              size_t ws_size, hipStream_t stream) {
  const float* x     = (const float*)d_in[0];
  const float* W1    = (const float*)d_in[1];
  const float* b1    = (const float*)d_in[2];
  const float* beta1 = (const float*)d_in[3];
  const float* thr1  = (const float*)d_in[4];
  const float* V1w   = (const float*)d_in[5];
  const float* V1b   = (const float*)d_in[6];
  const float* W2    = (const float*)d_in[7];
  const float* b2    = (const float*)d_in[8];
  const float* beta2 = (const float*)d_in[9];
  const float* thr2  = (const float*)d_in[10];
  const float* V2w   = (const float*)d_in[11];
  const float* V2b   = (const float*)d_in[12];
  const float* Wo    = (const float*)d_in[13];
  const float* bo    = (const float*)d_in[14];
  const float* beta3 = (const float*)d_in[15];
  const float* thr3  = (const float*)d_in[16];
  const float* Vow   = (const float*)d_in[17];
  const float* Vob   = (const float*)d_in[18];

  // ---- workspace layout (256B aligned)
  char* ws = (char*)d_ws;
  size_t off = 0;
  auto alloc = [&](size_t bytes) {
    size_t o = off;
    off = (off + bytes + 255) & ~(size_t)255;
    return o;
  };
  const int WP1_ELEMS = (( NIN + NHID) / 32) * (NHID / 16) * 512; // 5,242,880
  const int WP2_ELEMS = ((NHID + NHID) / 32) * (NHID / 16) * 512; // 8,388,608
  const int WPO_ELEMS = ((NHID + NOUT) / 32) * (NOUT / 16) * 512; //   278,528

  bf16_t* Wp1 = (bf16_t*)(ws + alloc((size_t)WP1_ELEMS * 2));
  bf16_t* Wp2 = (bf16_t*)(ws + alloc((size_t)WP2_ELEMS * 2));
  bf16_t* Wpo = (bf16_t*)(ws + alloc((size_t)WPO_ELEMS * 2));
  float*  bc1 = (float*)(ws + alloc(NHID * 4));
  float*  bc2 = (float*)(ws + alloc(NHID * 4));
  float*  bco = (float*)(ws + alloc(NOUT * 4));

  const size_t carry_begin = off;
  float*  m1     = (float*)(ws + alloc((size_t)BSZ * NHID * 4));
  float*  m2     = (float*)(ws + alloc((size_t)BSZ * NHID * 4));
  float*  mo     = (float*)(ws + alloc((size_t)BSZ * NOUT * 4));
  bf16_t* m2bf   = (bf16_t*)(ws + alloc((size_t)BSZ * NHID * 2));
  bf16_t* s1bf[2], *s2bf[2], *sobf[2];
  s1bf[0] = (bf16_t*)(ws + alloc((size_t)BSZ * NHID * 2));
  s1bf[1] = (bf16_t*)(ws + alloc((size_t)BSZ * NHID * 2));
  s2bf[0] = (bf16_t*)(ws + alloc((size_t)BSZ * NHID * 2));
  s2bf[1] = (bf16_t*)(ws + alloc((size_t)BSZ * NHID * 2));
  sobf[0] = (bf16_t*)(ws + alloc((size_t)BSZ * NOUT * 2));
  sobf[1] = (bf16_t*)(ws + alloc((size_t)BSZ * NOUT * 2));
  const size_t carry_bytes = off - carry_begin;

  // ---- zero carries (determinism across graph replays)
  {
    int n = (int)(carry_bytes / 4);
    zero_u32<<<(n + 255) / 256, 256, 0, stream>>>(
        (unsigned int*)(ws + carry_begin), n);
  }

  // ---- one-time weight packing + bias fusion (cheap; graph-captured)
  pack_w<<<(WP1_ELEMS + 255) / 256, 256, 0, stream>>>(W1, NIN,  V1w, NHID, NHID, Wp1, WP1_ELEMS);
  pack_w<<<(WP2_ELEMS + 255) / 256, 256, 0, stream>>>(W2, NHID, V2w, NHID, NHID, Wp2, WP2_ELEMS);
  pack_w<<<(WPO_ELEMS + 255) / 256, 256, 0, stream>>>(Wo, NHID, Vow, NOUT, NOUT, Wpo, WPO_ELEMS);
  combine_bias<<<(NHID + 255) / 256, 256, 0, stream>>>(b1, V1b, bc1, NHID);
  combine_bias<<<(NHID + 255) / 256, 256, 0, stream>>>(b2, V2b, bc2, NHID);
  combine_bias<<<(NOUT + 255) / 256, 256, 0, stream>>>(bo, Vob, bco, NOUT);

  float* spk = (float*)d_out;
  float* mem = spk + (size_t)T_STEPS * BSZ * NOUT;

  const dim3 blk(64);                   // 2 waves per block
  const dim3 gH(BSZ / 16, NHID / 128);  // (16,16) = 256 blocks
  const dim3 gO(BSZ / 16, NOUT / 128);  // (16,1)

  for (int t = 0; t < T_STEPS; ++t) {
    const int r = t & 1, w = r ^ 1;
    // layer 1: A = [x_t | s1_prev], K = 512 + 2048
    rleaky_gemm<0><<<gH, blk, 0, stream>>>(
        x + (size_t)t * BSZ * NIN, NIN, s1bf[r], NHID, Wp1, bc1, beta1, thr1,
        m1, s1bf[w], (bf16_t*)nullptr, (float*)nullptr, (float*)nullptr, NHID);
    // layer 2: A = [s1_new | s2_prev], K = 2048 + 2048
    rleaky_gemm<1><<<gH, blk, 0, stream>>>(
        s1bf[w], NHID, s2bf[r], NHID, Wp2, bc2, beta2, thr2,
        m2, s2bf[w], m2bf, (float*)nullptr, (float*)nullptr, NHID);
    // output: A = [m2 | so_prev], K = 2048 + 128, writes spk/mem records
    rleaky_gemm<2><<<gO, blk, 0, stream>>>(
        m2bf, NHID, sobf[r], NOUT, Wpo, bco, beta3, thr3,
        mo, sobf[w], (bf16_t*)nullptr,
        spk + (size_t)t * BSZ * NOUT, mem + (size_t)t * BSZ * NOUT, NOUT);
  }
}
